// TransformerBlock_66348654789219
// MI455X (gfx1250) — matrix operand order repaired and checked by reading, not compiled, not run on hardware
//
#include <hip/hip_runtime.h>
#include <math.h>

// ---------------------------------------------------------------------------
// Types
// ---------------------------------------------------------------------------
typedef __attribute__((ext_vector_type(4)))  unsigned int u32x4;
typedef __attribute__((ext_vector_type(4)))  int          v4i;
typedef __attribute__((ext_vector_type(8)))  float        v8f;
typedef __attribute__((ext_vector_type(16))) __bf16       v16bf;

struct Frag32 { u32x4 lo, hi; };

#if defined(__has_builtin)
#if __has_builtin(__builtin_amdgcn_global_load_async_to_lds_b128) && \
    __has_builtin(__builtin_amdgcn_s_wait_asynccnt)
#define USE_ASYNC 1
#endif
#endif
#ifndef USE_ASYNC
#define USE_ASYNC 0
#endif

__device__ __forceinline__ unsigned short f2bf(float f) {
  unsigned int u = __float_as_uint(f);
  u += 0x7fffu + ((u >> 16) & 1u);   // round-to-nearest-even
  return (unsigned short)(u >> 16);
}

__device__ __forceinline__ v16bf load_frag(const unsigned short* p0,
                                           const unsigned short* p1) {
  Frag32 fr;
  fr.lo = *(const u32x4*)p0;
  fr.hi = *(const u32x4*)p1;
  return __builtin_bit_cast(v16bf, fr);
}

#if USE_ASYNC
// 16B global -> LDS async copy (per lane), tracked by ASYNCcnt.
// Builtin prototype (from hipcc diagnostic): param0 = int4 addrspace(1)*,
// param1 = int4 addrspace(3)*, then imm offset, imm cpol.
__device__ __forceinline__ void async_cp16(const unsigned short* g,
                                           unsigned short* l) {
  __builtin_amdgcn_global_load_async_to_lds_b128(
      (__attribute__((address_space(1))) v4i*)g,
      (__attribute__((address_space(3))) v4i*)l,
      0 /*imm offset*/, 0 /*cpol*/);
}
#endif

// ---------------------------------------------------------------------------
// Generic bf16 WMMA GEMM:  C[m][n] = sum_k A[m][k] * W[n][k]   (A @ W^T)
// Block tile 128x128, K-step 32, 256 threads = 8 waves (4 M-waves x 2 N-waves)
// Each wave: 32x64 = 2x4 subtiles of 16x16, v_wmma_f32_16x16x32_bf16.
// Double-buffered LDS fed by GLOBAL_LOAD_ASYNC_TO_LDS_B128 when available.
// ---------------------------------------------------------------------------
#define MODE_BF16   0   // store bf16 C[m*ldc+n]
#define MODE_BF16T  1   // store bf16 C[n*ldc+m]  (transposed, for Vt)
#define MODE_GELU   2   // store bf16 gelu(acc)
#define MODE_F32    3   // store f32  acc*alpha   (scores)
#define MODE_F32ADD 4   // store f32  acc + Add[m*ldc+n]

#define TILE_USH (128 * 40)          // one padded 128x32 bf16 tile in LDS

__global__ __launch_bounds__(256)
void gemm_bf16_kernel(const unsigned short* __restrict__ A, long long sAb, int lda,
                      const unsigned short* __restrict__ W, long long sWb, int ldw,
                      void* __restrict__ Cv, long long sCb, int ldc,
                      const float* __restrict__ Add, long long sAddb,
                      int K, float alpha, int mode, int causal)
{
  if (causal && blockIdx.x > blockIdx.y) return;   // tile fully above diagonal

  extern __shared__ unsigned short sm[];
  // layout: As0 | Bs0 | As1 | Bs1   (each TILE_USH ushorts)

  const int tid  = threadIdx.x;
  const int lane = tid & 31;
  const int wave = tid >> 5;
  const int wm   = wave >> 1;   // 0..3 : M-wave
  const int wn   = wave & 1;    // 0..1 : N-wave
  const int r    = lane & 15;
  const int hh   = lane >> 4;   // half-wave select

  const long long m0 = (long long)blockIdx.y * 128;
  const long long n0 = (long long)blockIdx.x * 128;

  const unsigned short* Ab = A + (long long)blockIdx.z * sAb;
  const unsigned short* Wb = W + (long long)blockIdx.z * sWb;

  v8f acc[2][4] = {};

#if USE_ASYNC
  // ---- double-buffered async pipeline ----
  auto issue = [&](int buf, int k0) {
    unsigned short* Asb = sm + buf * 2 * TILE_USH;
    unsigned short* Bsb = Asb + TILE_USH;
#pragma unroll
    for (int c = tid; c < 512; c += 256) {       // 2 iters: 4 async loads/thread
      int row = c >> 2;
      int k8  = (c & 3) << 3;
      async_cp16(&Ab[(m0 + row) * (long long)lda + k0 + k8], &Asb[row * 40 + k8]);
      async_cp16(&Wb[(n0 + row) * (long long)ldw + k0 + k8], &Bsb[row * 40 + k8]);
    }
  };

  issue(0, 0);
  int buf = 0;
  for (int k0 = 0; k0 < K; k0 += 32) {
    const bool has_next = (k0 + 32 < K);
    if (has_next) issue(buf ^ 1, k0 + 32);       // prefetch into other buffer
    // async loads retire in order: leave the 4 newest in flight.
    if (has_next) __builtin_amdgcn_s_wait_asynccnt(4);
    else          __builtin_amdgcn_s_wait_asynccnt(0);
    __syncthreads();                             // all waves' current tile in LDS

    const unsigned short* As = sm + buf * 2 * TILE_USH;
    const unsigned short* Bs = As + TILE_USH;

    v16bf af[2], bfr[4];
#pragma unroll
    for (int mt = 0; mt < 2; ++mt) {
      const unsigned short* base = &As[(wm * 32 + mt * 16 + r) * 40 + 8 * hh];
      af[mt] = load_frag(base, base + 16);
    }
#pragma unroll
    for (int nt = 0; nt < 4; ++nt) {
      const unsigned short* base = &Bs[(wn * 64 + nt * 16 + r) * 40 + 8 * hh];
      bfr[nt] = load_frag(base, base + 16);
    }
#pragma unroll
    for (int mt = 0; mt < 2; ++mt)
#pragma unroll
      for (int nt = 0; nt < 4; ++nt)
        acc[mt][nt] = __builtin_amdgcn_wmma_f32_16x16x32_bf16(
            false, af[mt], false, bfr[nt], (short)0, acc[mt][nt], false, false);
    __syncthreads();                             // reads done before buf reuse
    buf ^= 1;
  }
#else
  // ---- synchronous fallback (single buffer) ----
  unsigned short* As = sm;
  unsigned short* Bs = sm + TILE_USH;
  for (int k0 = 0; k0 < K; k0 += 32) {
    for (int c = tid; c < 512; c += 256) {
      int row = c >> 2;
      int k8  = (c & 3) << 3;
      *(u32x4*)&As[row * 40 + k8] =
          *(const u32x4*)&Ab[(m0 + row) * (long long)lda + k0 + k8];
      *(u32x4*)&Bs[row * 40 + k8] =
          *(const u32x4*)&Wb[(n0 + row) * (long long)ldw + k0 + k8];
    }
    __syncthreads();
    v16bf af[2], bfr[4];
#pragma unroll
    for (int mt = 0; mt < 2; ++mt) {
      const unsigned short* base = &As[(wm * 32 + mt * 16 + r) * 40 + 8 * hh];
      af[mt] = load_frag(base, base + 16);
    }
#pragma unroll
    for (int nt = 0; nt < 4; ++nt) {
      const unsigned short* base = &Bs[(wn * 64 + nt * 16 + r) * 40 + 8 * hh];
      bfr[nt] = load_frag(base, base + 16);
    }
#pragma unroll
    for (int mt = 0; mt < 2; ++mt)
#pragma unroll
      for (int nt = 0; nt < 4; ++nt)
        acc[mt][nt] = __builtin_amdgcn_wmma_f32_16x16x32_bf16(
            false, af[mt], false, bfr[nt], (short)0, acc[mt][nt], false, false);
    __syncthreads();
  }
#endif

  // ---- epilogue ----
  float*          Cf   = (float*)Cv          + (long long)blockIdx.z * sCb;
  unsigned short* Cs   = (unsigned short*)Cv + (long long)blockIdx.z * sCb;
  const float*    Addb = Add ? Add + (long long)blockIdx.z * sAddb : nullptr;

#pragma unroll
  for (int mt = 0; mt < 2; ++mt) {
#pragma unroll
    for (int nt = 0; nt < 4; ++nt) {
      long long n     = n0 + wn * 64 + nt * 16 + r;
      long long mbase = m0 + wm * 32 + mt * 16 + 8 * hh;
      v8f a = acc[mt][nt];
#pragma unroll
      for (int v = 0; v < 8; ++v) {
        long long m = mbase + v;
        float val = a[v];
        if (mode == MODE_BF16) {
          Cs[m * ldc + n] = f2bf(val);
        } else if (mode == MODE_BF16T) {
          Cs[n * ldc + m] = f2bf(val);
        } else if (mode == MODE_GELU) {
          Cs[m * ldc + n] = f2bf(0.5f * val * (1.0f + erff(val * 0.7071067811865476f)));
        } else if (mode == MODE_F32) {
          Cf[m * ldc + n] = val * alpha;
        } else { // MODE_F32ADD
          Cf[m * ldc + n] = val + Addb[m * ldc + n];
        }
      }
    }
  }
}

// ---------------------------------------------------------------------------
// Fused dual RMSNorm: h1 = rms(x)*ln1, h2 = rms(x)*ln2  (bf16 out), D=1024
// ---------------------------------------------------------------------------
__global__ __launch_bounds__(256)
void rmsnorm2_kernel(const float* __restrict__ x,
                     const float* __restrict__ g1, const float* __restrict__ g2,
                     unsigned short* __restrict__ h1, unsigned short* __restrict__ h2)
{
  const int D = 1024;
  size_t row = blockIdx.x;
  const float* xr = x + row * D;
  float ss = 0.f;
  for (int i = threadIdx.x; i < D; i += 256) { float v = xr[i]; ss += v * v; }
  for (int o = 16; o; o >>= 1) ss += __shfl_xor(ss, o, 32);
  __shared__ float sh[8];
  int lane = threadIdx.x & 31, w = threadIdx.x >> 5;
  if (lane == 0) sh[w] = ss;
  __syncthreads();
  float tot = 0.f;
  for (int i = 0; i < 8; ++i) tot += sh[i];
  float inv = rsqrtf(tot / (float)D + 1e-5f);
  for (int i = threadIdx.x; i < D; i += 256) {
    float v = xr[i] * inv;
    h1[row * D + i] = f2bf(v * g1[i]);
    h2[row * D + i] = f2bf(v * g2[i]);
  }
}

// ---------------------------------------------------------------------------
// Causal softmax over fp32 scores row -> bf16 probs (zeros above diagonal)
// ---------------------------------------------------------------------------
__global__ __launch_bounds__(256)
void causal_softmax_kernel(const float* __restrict__ S,
                           unsigned short* __restrict__ P, int L)
{
  size_t row = blockIdx.x;            // over B*S rows; [B,S,S] is contiguous
  int i = (int)(row % (size_t)L);
  const float* sr = S + row * (size_t)L;
  unsigned short* pr = P + row * (size_t)L;
  __shared__ float sh[8];
  int lane = threadIdx.x & 31, w = threadIdx.x >> 5;

  float mx = -3.0e38f;
  for (int j = threadIdx.x; j <= i; j += 256) mx = fmaxf(mx, sr[j]);
  for (int o = 16; o; o >>= 1) mx = fmaxf(mx, __shfl_xor(mx, o, 32));
  if (lane == 0) sh[w] = mx;
  __syncthreads();
  mx = sh[0];
  for (int k = 1; k < 8; ++k) mx = fmaxf(mx, sh[k]);
  __syncthreads();

  float se = 0.f;
  for (int j = threadIdx.x; j <= i; j += 256) se += expf(sr[j] - mx);
  for (int o = 16; o; o >>= 1) se += __shfl_xor(se, o, 32);
  if (lane == 0) sh[w] = se;
  __syncthreads();
  float tot = 0.f;
  for (int k = 0; k < 8; ++k) tot += sh[k];
  float invs = 1.0f / tot;

  for (int j = threadIdx.x; j < L; j += 256) {
    float p = 0.f;
    if (j <= i) p = expf(sr[j] - mx) * invs;
    pr[j] = f2bf(p);
  }
}

// ---------------------------------------------------------------------------
// fp32 -> bf16 converts (plain and transposed, for weights)
// ---------------------------------------------------------------------------
__global__ void cvt_kernel(const float* __restrict__ s,
                           unsigned short* __restrict__ d, size_t n)
{
  for (size_t i = (size_t)blockIdx.x * blockDim.x + threadIdx.x; i < n;
       i += (size_t)gridDim.x * blockDim.x)
    d[i] = f2bf(s[i]);
}

__global__ void cvtT_kernel(const float* __restrict__ s,
                            unsigned short* __restrict__ d, int rows, int cols)
{
  size_t n = (size_t)rows * cols;
  for (size_t i = (size_t)blockIdx.x * blockDim.x + threadIdx.x; i < n;
       i += (size_t)gridDim.x * blockDim.x) {
    size_t rr = i / (size_t)cols, cc = i % (size_t)cols;
    d[cc * (size_t)rows + rr] = f2bf(s[i]);
  }
}

// ---------------------------------------------------------------------------
// Host orchestration
// ---------------------------------------------------------------------------
extern "C" void kernel_launch(void* const* d_in, const int* in_sizes, int n_in,
                              void* d_out, int out_size, void* d_ws, size_t ws_size,
                              hipStream_t stream)
{
  (void)in_sizes; (void)n_in; (void)out_size; (void)ws_size;
  const float* x        = (const float*)d_in[0];
  const float* ln1      = (const float*)d_in[1];
  const float* ln2      = (const float*)d_in[2];
  const float* ffn1     = (const float*)d_in[3];
  const float* ffn2     = (const float*)d_in[4];
  const float* k_proj   = (const float*)d_in[5];
  const float* q_proj   = (const float*)d_in[6];
  const float* v_proj   = (const float*)d_in[7];
  const float* out_proj = (const float*)d_in[8];
  float* out = (float*)d_out;

  constexpr int  Bb = 4, S = 2048, D = 1024, F = 4096;
  constexpr long long BS = (long long)Bb * S;   // 8192 rows
  const size_t MB = 1ull << 20;
  char* ws = (char*)d_ws;

  unsigned short* h1  = (unsigned short*)(ws + 0 * MB);    // 16MB
  unsigned short* h2  = (unsigned short*)(ws + 16 * MB);   // 16MB
  unsigned short* wq  = (unsigned short*)(ws + 32 * MB);   // 2MB
  unsigned short* wk  = (unsigned short*)(ws + 34 * MB);   // 2MB
  unsigned short* wv  = (unsigned short*)(ws + 36 * MB);   // 2MB
  unsigned short* woT = (unsigned short*)(ws + 38 * MB);   // 2MB
  unsigned short* w1  = (unsigned short*)(ws + 40 * MB);   // 8MB
  unsigned short* w2  = (unsigned short*)(ws + 48 * MB);   // 8MB
  unsigned short* Qb  = (unsigned short*)(ws + 56 * MB);   // 16MB
  unsigned short* Kb  = (unsigned short*)(ws + 72 * MB);   // 16MB
  unsigned short* Vt  = (unsigned short*)(ws + 88 * MB);   // 16MB  [B][D][S]
  unsigned short* P   = (unsigned short*)(ws + 104 * MB);  // 32MB  [B][S][S]
  unsigned short* ctx = (unsigned short*)(ws + 136 * MB);  // 16MB
  float*          Sc  = (float*)(ws + 152 * MB);           // 64MB  scores fp32
  unsigned short* ffh = (unsigned short*)(ws + 152 * MB);  // reuse: 64MB bf16

  const size_t shmem = 4 * TILE_USH * sizeof(unsigned short); // 40960 B

  auto gemm = [&](const unsigned short* A, long long sAb, int lda,
                  const unsigned short* W, long long sWb, int ldw,
                  void* C, long long sCb, int ldc,
                  const float* Add, long long sAddb,
                  int M, int N, int K, float alpha, int mode, int causal, int nb) {
    dim3 grid(N / 128, M / 128, nb);
    gemm_bf16_kernel<<<grid, 256, shmem, stream>>>(
        A, sAb, lda, W, sWb, ldw, C, sCb, ldc, Add, sAddb, K, alpha, mode, causal);
  };

  // 1) dual RMSNorm
  rmsnorm2_kernel<<<(unsigned)BS, 256, 0, stream>>>(x, ln1, ln2, h1, h2);

  // 2) weight conversion to bf16 (out_proj transposed)
  cvt_kernel <<<2048, 256, 0, stream>>>(q_proj, wq, (size_t)D * D);
  cvt_kernel <<<2048, 256, 0, stream>>>(k_proj, wk, (size_t)D * D);
  cvt_kernel <<<2048, 256, 0, stream>>>(v_proj, wv, (size_t)D * D);
  cvtT_kernel<<<2048, 256, 0, stream>>>(out_proj, woT, D, D);
  cvt_kernel <<<4096, 256, 0, stream>>>(ffn1, w1, (size_t)F * D);
  cvt_kernel <<<4096, 256, 0, stream>>>(ffn2, w2, (size_t)D * F);

  // 3) Q = h1 @ q_proj^T ; K = h1 @ k_proj^T ; Vt = (h1 @ v_proj^T)^T per batch
  gemm(h1, 0, D, wq, 0, D, Qb, 0, D, nullptr, 0, (int)BS, D, D, 1.f, MODE_BF16, 0, 1);
  gemm(h1, 0, D, wk, 0, D, Kb, 0, D, nullptr, 0, (int)BS, D, D, 1.f, MODE_BF16, 0, 1);
  gemm(h1, (long long)S * D, D, wv, 0, D, Vt, (long long)D * S, S,
       nullptr, 0, S, D, D, 1.f, MODE_BF16T, 0, Bb);

  // 4) scores = Q @ K^T / 32 (batched, causal tile skip), fp32
  gemm(Qb, (long long)S * D, D, Kb, (long long)S * D, D, Sc, (long long)S * S, S,
       nullptr, 0, S, S, D, 0.03125f, MODE_F32, 1, Bb);

  // 5) causal softmax -> bf16 P (zeros above diagonal)
  causal_softmax_kernel<<<(unsigned)BS, 256, 0, stream>>>(Sc, P, S);

  // 6) ctx = P @ V  (B-operand = Vt rows, contiguous in K)
  gemm(P, (long long)S * S, S, Vt, (long long)D * S, S, ctx, (long long)S * D, D,
       nullptr, 0, S, D, S, 1.f, MODE_BF16, 0, Bb);

  // 7) y = x + ctx @ out_proj   -> d_out (fp32)
  gemm(ctx, 0, D, woT, 0, D, out, 0, D, x, 0, (int)BS, D, D, 1.f, MODE_F32ADD, 0, 1);

  // 8) ffh = gelu(h2 @ ffn1^T)  (bf16, reuses scores buffer)
  gemm(h2, 0, D, w1, 0, D, ffh, 0, F, nullptr, 0, (int)BS, F, D, 1.f, MODE_GELU, 0, 1);

  // 9) out = y + ffh @ ffn2^T
  gemm(ffh, 0, F, w2, 0, F, out, 0, D, out, 0, (int)BS, D, F, 1.f, MODE_F32ADD, 0, 1);
}
